// NoCrossLayerLight_51170240364936
// MI455X (gfx1250) — matrix operand order repaired and checked by reading, not compiled, not run on hardware
//
#include <hip/hip_runtime.h>
#include <hip/hip_bf16.h>

// ---------------------------------------------------------------------------
// NoCrossLayerLight fused pipeline for MI455X (gfx1250, wave32, WMMA + TDM)
//   B=8, N1=N2=4096, CIN=64, C0=128, K=16
// Pipeline:
//   0) cvt   : w_m1/w_m2 f32 -> f16 once into workspace
//   1) conv1 : p = feat^T * W^T + b  (WMMA f16, f32 accum) -> p1 (f32, bias
//              includes b_pos), p2 (f16)
//   2) knn   : per-query top-16 neighbor indices (register top-K + wave merge)
//   3) fused : TDM weight load to LDS + gather + pos-emb + 2x(128x128 WMMA MLP)
//              + max over K
// Workspace: p1 16MB + p2 8MB + idx 2MB + w_h 64KB  (~26.1MB of d_ws)
// ---------------------------------------------------------------------------

#define BATCH 8
#define NPTS  4096
#define CINC  64
#define C0C   128
#define KNN_K 16
#define LEAK  0.1f

typedef _Float16 v8h  __attribute__((ext_vector_type(8)));
typedef _Float16 v16h __attribute__((ext_vector_type(16)));
typedef float    v8f  __attribute__((ext_vector_type(8)));
typedef float    v4f  __attribute__((ext_vector_type(4)));
typedef unsigned v4u  __attribute__((ext_vector_type(4)));
typedef int      v4i  __attribute__((ext_vector_type(4)));
typedef int      v8i  __attribute__((ext_vector_type(8)));

// 0 < LEAK < 1  =>  leaky(x) == max(x, LEAK*x): 2 VALU ops, no cmp/cndmask
__device__ __forceinline__ float leaky(float x) { return fmaxf(x, LEAK * x); }

// A-fragment K-offset map (16-bit A 16x32, wave32):
//   low lanes (0-15):  elems 0..7 -> K 0..7,  elems 8..15 -> K 16..23
//   high lanes (16-31):elems 0..7 -> K 8..15, elems 8..15 -> K 24..31
__device__ __forceinline__ int a_koff(int e, int hi) {
    return ((e < 8) ? e : (e + 8)) + 8 * hi;
}

// ---------------------------------------------------------------------------
// Kernel 0: convert MLP weights to f16 once (tiny)
// ---------------------------------------------------------------------------
__global__ void __launch_bounds__(256)
cvt_weights_kernel(const float* __restrict__ w1, const float* __restrict__ w2,
                   _Float16* __restrict__ o1, _Float16* __restrict__ o2) {
    int i = blockIdx.x * 256 + threadIdx.x;
    if (i < C0C * C0C) {
        o1[i] = (_Float16)w1[i];
        o2[i] = (_Float16)w2[i];
    }
}

// ---------------------------------------------------------------------------
// Kernel 1: p[b,n,o] = sum_c feat[b,c,n] * w[o,c] + bias[o] (+ bias2[o])
// One wave computes a 16(point) x 128(channel) tile. blockDim = 128 (4 waves).
// ---------------------------------------------------------------------------
template <int STOREHALF>
__global__ void __launch_bounds__(128)
conv1_kernel(const float* __restrict__ feat, const float* __restrict__ w,
             const float* __restrict__ bias, const float* __restrict__ bias2,
             float* __restrict__ out_f32, _Float16* __restrict__ out_f16, int N) {
    __shared__ __align__(16) _Float16 Wl[C0C][72];   // padded rows (144B stride)
    __shared__ float Bl[C0C];

    int tid = threadIdx.x;
    for (int i = tid; i < C0C * CINC; i += 128) {
        int o = i / CINC, c = i % CINC;
        Wl[o][c] = (_Float16)w[i];
    }
    if (tid < C0C) {
        float bv = bias[tid];
        if (bias2) bv += bias2[tid];     // fold b_pos into p1 bias
        Bl[tid] = bv;
    }
    __syncthreads();

    int wv = tid >> 5, lane = tid & 31;
    int m = lane & 15, hi = lane >> 4;
    int gw = blockIdx.x * 4 + wv;
    int wavesPerB = N / 16;
    int b  = gw / wavesPerB;
    int n0 = (gw % wavesPerB) * 16;

    // A fragments: A[m][c] = feat[b][c][n0+m]  (K = CIN = 64 -> 2 blocks)
    v16h a[2];
    const float* fbase = feat + (size_t)b * CINC * N + n0 + m;
#pragma unroll
    for (int kb = 0; kb < 2; ++kb) {
#pragma unroll
        for (int e = 0; e < 16; ++e) {
            int c = kb * 32 + a_koff(e, hi);
            a[kb][e] = (_Float16)fbase[(size_t)c * N];
        }
    }

#pragma unroll
    for (int nt = 0; nt < 8; ++nt) {
        int o = nt * 16 + m;                 // output channel (= B column)
        // preload both B fragments, then run the WMMA chain
        v16h bf[2];
#pragma unroll
        for (int kb = 0; kb < 2; ++kb) {
            const v8h* wp = (const v8h*)&Wl[o][kb * 32 + hi * 8];
            v8h lo = wp[0], hp = wp[2];
#pragma unroll
            for (int t = 0; t < 8; ++t) { bf[kb][t] = lo[t]; bf[kb][8 + t] = hp[t]; }
        }
        float bv = Bl[o];
        v8f acc = {bv, bv, bv, bv, bv, bv, bv, bv};
#pragma unroll
        for (int kb = 0; kb < 2; ++kb)
            acc = __builtin_amdgcn_wmma_f32_16x16x32_f16(
                false, a[kb], false, bf[kb], (short)0, acc, false, false);
        // D layout: lane col = lane&15, VGPR r -> row r + 8*hi
        size_t base = ((size_t)b * N + n0 + 8 * hi) * C0C + o;
        if (STOREHALF) {
            _Float16* bp = out_f16 + base;
#pragma unroll
            for (int r = 0; r < 8; ++r) bp[r * C0C] = (_Float16)acc[r];
        } else {
            float* bp = out_f32 + base;
#pragma unroll
            for (int r = 0; r < 8; ++r) bp[r * C0C] = acc[r];
        }
    }
}

// ---------------------------------------------------------------------------
// Kernel 2: KNN (top-16 smallest distances). One wave per query.
// ---------------------------------------------------------------------------
__global__ void __launch_bounds__(256)
knn_kernel(const float* __restrict__ pc1, const float* __restrict__ pc2,
           int* __restrict__ idx) {
    __shared__ float md[8][512];
    __shared__ int   mi[8][512];

    int tid = threadIdx.x, wv = tid >> 5, lane = tid & 31;
    int q = blockIdx.x * 8 + wv;
    int b = q / NPTS, n = q % NPTS;

    const float* p2b = pc2 + (size_t)b * 3 * NPTS;
    float qx = pc1[(size_t)b * 3 * NPTS + 0 * NPTS + n];
    float qy = pc1[(size_t)b * 3 * NPTS + 1 * NPTS + n];
    float qz = pc1[(size_t)b * 3 * NPTS + 2 * NPTS + n];

    float bd[KNN_K]; int bi[KNN_K];
#pragma unroll
    for (int j = 0; j < KNN_K; ++j) { bd[j] = 3.4e38f; bi[j] = 0; }
    float worst = 3.4e38f;

    for (int mbase = 0; mbase < NPTS; mbase += 32) {
        int mm = mbase + lane;
        float dx = p2b[mm] - qx;
        float dy = p2b[NPTS + mm] - qy;
        float dz = p2b[2 * NPTS + mm] - qz;
        float d = dx * dx + dy * dy + dz * dz;
        if (d < worst) {
            int wp = 0; float wvv = bd[0];
#pragma unroll
            for (int j = 1; j < KNN_K; ++j) if (bd[j] > wvv) { wvv = bd[j]; wp = j; }
#pragma unroll
            for (int j = 0; j < KNN_K; ++j) if (j == wp) { bd[j] = d; bi[j] = mm; }
            wvv = bd[0];
#pragma unroll
            for (int j = 1; j < KNN_K; ++j) wvv = fmaxf(wvv, bd[j]);
            worst = wvv;
        }
    }

#pragma unroll
    for (int j = 0; j < KNN_K; ++j) {
        md[wv][lane * KNN_K + j] = bd[j];
        mi[wv][lane * KNN_K + j] = bi[j];
    }
    __syncthreads();

    for (int r = 0; r < KNN_K; ++r) {
        float v = 3.4e38f; int p = 0;
#pragma unroll
        for (int t = 0; t < 16; ++t) {
            int e = lane + 32 * t;
            float dv = md[wv][e];
            if (dv < v) { v = dv; p = e; }
        }
#pragma unroll
        for (int off = 16; off >= 1; off >>= 1) {
            float ov = __shfl_xor(v, off);
            int   op = __shfl_xor(p, off);
            if (ov < v || (ov == v && op < p)) { v = ov; p = op; }
        }
        if (lane == 0) {
            idx[(size_t)q * KNN_K + r] = mi[wv][p];
            md[wv][p] = 3.4e38f;
        }
        __syncthreads();
    }
}

// ---------------------------------------------------------------------------
// TDM: load one 128x128 f16 matrix from global into LDS with padded rows
// (row = 128 halves = 256B = 64 dwords, pad +4 dwords -> 136-half stride).
// ---------------------------------------------------------------------------
#if defined(__has_builtin)
#if __has_builtin(__builtin_amdgcn_tensor_load_to_lds) && __has_builtin(__builtin_amdgcn_s_wait_tensorcnt)
#define HAVE_TDM 1
#endif
#endif

#ifdef HAVE_TDM
__device__ __forceinline__ void tdm_load_weight_128x128_f16(const _Float16* gsrc,
                                                            _Float16* ldst) {
    unsigned long long ga = (unsigned long long)(size_t)gsrc;
    unsigned lds = (unsigned)(size_t)(void*)ldst;       // low 32 bits = LDS offset
    v4u g0;
    g0[0] = 1u;                                          // count=1, user descriptor
    g0[1] = lds;                                         // lds_addr
    g0[2] = (unsigned)(ga & 0xffffffffu);                // global_addr[31:0]
    g0[3] = (unsigned)((ga >> 32) & 0x1ffffffu)          // global_addr[56:32]
          | (2u << 30);                                  // type = 2 ("image")
    v8i g1;
    g1[0] = (1 << 16)                                    // data_size = 2 bytes
          | (1 << 20)                                    // pad_enable
          | (5 << 22)                                    // pad_interval: 64 dwords
          | (3 << 25);                                   // pad_amount: 4 dwords
    g1[1] = (C0C & 0xffff) << 16;                        // tensor_dim0 = 128 (lo16)
    g1[2] = (C0C & 0xffff) << 16;                        // tensor_dim1 = 128 (lo16)
    g1[3] = (C0C & 0xffff) << 16;                        // tile_dim0 = 128
    g1[4] = C0C;                                         // tile_dim1 = 128, tile_dim2 = 0
    g1[5] = C0C;                                         // tensor_dim0_stride = 128
    g1[6] = 0;
    g1[7] = 0;
    v4i z4 = {0, 0, 0, 0};
#if defined(__clang_major__) && __clang_major__ >= 23
    v8i z8 = {0, 0, 0, 0, 0, 0, 0, 0};
    __builtin_amdgcn_tensor_load_to_lds(g0, g1, z4, z4, z8, 0);
#else
    __builtin_amdgcn_tensor_load_to_lds(g0, g1, z4, z4, 0);
#endif
}
#endif

// ---------------------------------------------------------------------------
// Kernel 3: fused gather + pos_emb + leaky + MLP(128->128)x2 (WMMA) + max-K.
// One wave per query point (16 neighbor rows == one WMMA M-block).
// blockDim = 128 (4 waves / 4 points per block). Dynamic LDS ~87.5KB.
// ---------------------------------------------------------------------------
#define WPAD 136                      // half-stride: 272B = 68 dwords (4n banks)
#define LDS_W1   0
#define LDS_W2   (C0C * WPAD)                    // halves
#define LDS_B1   (2 * C0C * WPAD * 2)            // bytes from base
#define LDS_B2   (LDS_B1 + C0C * 4)
#define LDS_WPX  (LDS_B2 + C0C * 4)              // SoA pos weights (x,y,z)
#define LDS_WPY  (LDS_WPX + C0C * 4)
#define LDS_WPZ  (LDS_WPY + C0C * 4)
#define LDS_HB   (LDS_WPZ + C0C * 4)             // 4 waves * 16 * WPAD halves
#define LDS_TOTAL (LDS_HB + 4 * 16 * WPAD * 2)

__global__ void __launch_bounds__(128)
fused_mlp_kernel(const float* __restrict__ p1, const _Float16* __restrict__ p2,
                 const int* __restrict__ idx,
                 const float* __restrict__ pc1, const float* __restrict__ pc2,
                 const float* __restrict__ w_pos,
                 const _Float16* __restrict__ wm1h, const float* __restrict__ b_m1,
                 const _Float16* __restrict__ wm2h, const float* __restrict__ b_m2,
                 float* __restrict__ out) {
    extern __shared__ __align__(16) char smem[];
    _Float16* W1  = (_Float16*)smem + LDS_W1;
    _Float16* W2  = (_Float16*)smem + LDS_W2;
    float*    B1  = (float*)(smem + LDS_B1);
    float*    B2  = (float*)(smem + LDS_B2);
    float*    WPx = (float*)(smem + LDS_WPX);
    float*    WPy = (float*)(smem + LDS_WPY);
    float*    WPz = (float*)(smem + LDS_WPZ);
    _Float16* HB  = (_Float16*)(smem + LDS_HB);

    int tid = threadIdx.x;

#ifdef HAVE_TDM
    if (tid < 32) {     // wave 0 issues both tensor DMAs
        tdm_load_weight_128x128_f16(wm1h, W1);
        tdm_load_weight_128x128_f16(wm2h, W2);
        __builtin_amdgcn_s_wait_tensorcnt(0);
    }
#else
    for (int i = tid; i < C0C * (C0C / 8); i += 128) {   // v8h rows, 16 chunks/row
        int o = i >> 4, ch = i & 15;
        *(v8h*)(W1 + o * WPAD + ch * 8) = *(const v8h*)(wm1h + o * C0C + ch * 8);
        *(v8h*)(W2 + o * WPAD + ch * 8) = *(const v8h*)(wm2h + o * C0C + ch * 8);
    }
#endif
    for (int i = tid; i < C0C; i += 128) {
        B1[i]  = b_m1[i];
        B2[i]  = b_m2[i];
        WPx[i] = w_pos[i * 3 + 0];
        WPy[i] = w_pos[i * 3 + 1];
        WPz[i] = w_pos[i * 3 + 2];
    }
    __syncthreads();

    int wv = tid >> 5, lane = tid & 31, m = lane & 15, hi = lane >> 4;
    int pidx = blockIdx.x * 4 + wv;
    int b = pidx / NPTS, n = pidx % NPTS;

    int j = idx[(size_t)pidx * KNN_K + m];       // neighbor row for this lane
    float qx = pc1[(size_t)b * 3 * NPTS + n];
    float qy = pc1[(size_t)b * 3 * NPTS + NPTS + n];
    float qz = pc1[(size_t)b * 3 * NPTS + 2 * NPTS + n];
    float dx = pc2[(size_t)b * 3 * NPTS + j] - qx;
    float dy = pc2[(size_t)b * 3 * NPTS + NPTS + j] - qy;
    float dz = pc2[(size_t)b * 3 * NPTS + 2 * NPTS + j] - qz;

    const _Float16* p2row = p2 + ((size_t)b * NPTS + j) * C0C;
    const float*    p1row = p1 + ((size_t)b * NPTS + n) * C0C;   // includes b_pos

    // h0 = leaky(p2[j] + p1[n] + dir . w_pos)  built directly as A fragments
    v16h a0[4];
#pragma unroll
    for (int kb = 0; kb < 4; ++kb) {
        int c0 = kb * 32 + 8 * hi;
        v8h plo = *(const v8h*)(p2row + c0);
        v8h phi = *(const v8h*)(p2row + c0 + 16);
        v4f p1v4[4] = { *(const v4f*)(p1row + c0),      *(const v4f*)(p1row + c0 + 4),
                        *(const v4f*)(p1row + c0 + 16), *(const v4f*)(p1row + c0 + 20) };
        v4f wx4[4]  = { *(const v4f*)(WPx + c0),        *(const v4f*)(WPx + c0 + 4),
                        *(const v4f*)(WPx + c0 + 16),   *(const v4f*)(WPx + c0 + 20) };
        v4f wy4[4]  = { *(const v4f*)(WPy + c0),        *(const v4f*)(WPy + c0 + 4),
                        *(const v4f*)(WPy + c0 + 16),   *(const v4f*)(WPy + c0 + 20) };
        v4f wz4[4]  = { *(const v4f*)(WPz + c0),        *(const v4f*)(WPz + c0 + 4),
                        *(const v4f*)(WPz + c0 + 16),   *(const v4f*)(WPz + c0 + 20) };
#pragma unroll
        for (int e = 0; e < 16; ++e) {
            int q = e >> 2, s = e & 3;              // chunk select (constant-folded)
            float pv = (e < 8) ? (float)plo[e] : (float)phi[e - 8];
            float v = pv + p1v4[q][s]
                    + dx * wx4[q][s] + dy * wy4[q][s] + dz * wz4[q][s];
            a0[kb][e] = (_Float16)leaky(v);
        }
    }

    _Float16* hb = HB + wv * 16 * WPAD;

    // ---- layer 1: h1 = leaky(h0 @ W1^T + b1), spill to LDS row-major ----
#pragma unroll
    for (int nt = 0; nt < 8; ++nt) {
        int o = nt * 16 + m;
        v16h bf[4];
#pragma unroll
        for (int kb = 0; kb < 4; ++kb) {
            const v8h* wp = (const v8h*)(W1 + o * WPAD + kb * 32 + hi * 8);
            v8h lo = wp[0], hp = wp[2];
#pragma unroll
            for (int t = 0; t < 8; ++t) { bf[kb][t] = lo[t]; bf[kb][8 + t] = hp[t]; }
        }
        float bv = B1[o];
        v8f acc = {bv, bv, bv, bv, bv, bv, bv, bv};
#pragma unroll
        for (int kb = 0; kb < 4; ++kb)
            acc = __builtin_amdgcn_wmma_f32_16x16x32_f16(
                false, a0[kb], false, bf[kb], (short)0, acc, false, false);
#pragma unroll
        for (int r = 0; r < 8; ++r)
            hb[(r + 8 * hi) * WPAD + o] = (_Float16)leaky(acc[r]);
    }
    __syncthreads();   // D-layout stores -> A-layout loads (cross-lane in wave)

    // reload h1 as A fragments
    v16h a1[4];
#pragma unroll
    for (int kb = 0; kb < 4; ++kb) {
        const v8h* hp0 = (const v8h*)(hb + m * WPAD + kb * 32 + 8 * hi);
        const v8h* hp1 = (const v8h*)(hb + m * WPAD + kb * 32 + 16 + 8 * hi);
        v8h lo = *hp0, hp = *hp1;
#pragma unroll
        for (int t = 0; t < 8; ++t) { a1[kb][t] = lo[t]; a1[kb][8 + t] = hp[t]; }
    }

    // ---- layer 2 + max over the K(=row) axis ----
#pragma unroll
    for (int nt = 0; nt < 8; ++nt) {
        int o = nt * 16 + m;
        v16h bf[4];
#pragma unroll
        for (int kb = 0; kb < 4; ++kb) {
            const v8h* wp = (const v8h*)(W2 + o * WPAD + kb * 32 + hi * 8);
            v8h lo = wp[0], hp = wp[2];
#pragma unroll
            for (int t = 0; t < 8; ++t) { bf[kb][t] = lo[t]; bf[kb][8 + t] = hp[t]; }
        }
        float bv = B2[o];
        v8f acc = {bv, bv, bv, bv, bv, bv, bv, bv};
#pragma unroll
        for (int kb = 0; kb < 4; ++kb)
            acc = __builtin_amdgcn_wmma_f32_16x16x32_f16(
                false, a1[kb], false, bf[kb], (short)0, acc, false, false);
        float vm = -3.4e38f;
#pragma unroll
        for (int r = 0; r < 8; ++r) vm = fmaxf(vm, leaky(acc[r]));
        vm = fmaxf(vm, __shfl_xor(vm, 16));   // rows 0-7 vs rows 8-15
        if (lane < 16)
            out[((size_t)b * C0C + o) * NPTS + n] = vm;   // [B, C, N1]
    }
}

// ---------------------------------------------------------------------------
extern "C" void kernel_launch(void* const* d_in, const int* in_sizes, int n_in,
                              void* d_out, int out_size, void* d_ws, size_t ws_size,
                              hipStream_t stream) {
    const float* pc1   = (const float*)d_in[0];
    const float* pc2   = (const float*)d_in[1];
    const float* feat1 = (const float*)d_in[2];
    const float* feat2 = (const float*)d_in[3];
    const float* w_t1  = (const float*)d_in[4];
    const float* b_t1  = (const float*)d_in[5];
    const float* w_t2  = (const float*)d_in[6];
    const float* b_t2  = (const float*)d_in[7];
    const float* w_pos = (const float*)d_in[8];
    const float* b_pos = (const float*)d_in[9];
    const float* w_m1  = (const float*)d_in[10];
    const float* b_m1  = (const float*)d_in[11];
    const float* w_m2  = (const float*)d_in[12];
    const float* b_m2  = (const float*)d_in[13];
    float* out = (float*)d_out;

    char* ws = (char*)d_ws;
    size_t p1Bytes  = (size_t)BATCH * NPTS * C0C * sizeof(float);     // 16 MB
    size_t p2Bytes  = (size_t)BATCH * NPTS * C0C * sizeof(_Float16);  //  8 MB
    size_t idxBytes = (size_t)BATCH * NPTS * KNN_K * sizeof(int);     //  2 MB
    float*    p1b    = (float*)ws;
    _Float16* p2b    = (_Float16*)(ws + p1Bytes);
    int*      idxbuf = (int*)(ws + p1Bytes + p2Bytes);
    _Float16* wm1h   = (_Float16*)(ws + p1Bytes + p2Bytes + idxBytes);
    _Float16* wm2h   = wm1h + C0C * C0C;

    // 0) one-time f32 -> f16 weight conversion (16K elems each)
    cvt_weights_kernel<<<(C0C * C0C + 255) / 256, 256, 0, stream>>>(
        w_m1, w_m2, wm1h, wm2h);

    // 1) p1 (f32, bias folds b_pos) and p2 (f16): 512 blocks x 128 threads
    conv1_kernel<0><<<BATCH * NPTS / 64, 128, 0, stream>>>(
        feat1, w_t1, b_t1, b_pos, p1b, (_Float16*)nullptr, NPTS);
    conv1_kernel<1><<<BATCH * NPTS / 64, 128, 0, stream>>>(
        feat2, w_t2, b_t2, (const float*)nullptr, (float*)nullptr, p2b, NPTS);

    // 2) KNN: 8 queries per block
    knn_kernel<<<BATCH * NPTS / 8, 256, 0, stream>>>(pc1, pc2, idxbuf);

    // 3) fused gather + MLP + maxpool: 4 points per block, ~87.5KB dynamic LDS
    fused_mlp_kernel<<<BATCH * NPTS / 4, 128, LDS_TOTAL, stream>>>(
        p1b, p2b, idxbuf, pc1, pc2, w_pos,
        wm1h, b_m1, wm2h, b_m2, out);
}
